// PointDecoder3_14070312861808
// MI455X (gfx1250) — compile-verified, matrix operand order checked by reading, not run on hardware
//
#include <hip/hip_runtime.h>
#include <hip/hip_bf16.h>
#include <stddef.h>

// ---------------- model constants ----------------
constexpr int BS   = 32;
constexpr int NIN  = 2048;
constexpr int PART = 4;
constexpr int HID  = 256;
constexpr int NH   = 8;
constexpr int HD   = 32;
constexpr int FFD  = 1024;
constexpr int NL   = 4;
constexpr int P    = 1024;   // POINT_NUM
constexpr int SEQ  = 1024;   // == P
constexpr int MAX_TOKEN = 6;

typedef __attribute__((ext_vector_type(16))) __bf16 v16bf;
typedef __attribute__((ext_vector_type(8)))  float  v8f;

#define DEV __device__ __forceinline__

// ---------------- bf16 helpers (bit-exact, no __bf16 arithmetic) ----------------
DEV float bf2f(unsigned short s) {
    union { unsigned int u; float f; } c; c.u = ((unsigned int)s) << 16; return c.f;
}
DEV unsigned short f2bfbits(float f) {
    union { float f; unsigned int u; } c; c.f = f;
    unsigned int u = c.u;
    return (unsigned short)((u + 0x7fffu + ((u >> 16) & 1u)) >> 16);   // RNE
}
DEV __bf16 bfbits(unsigned short s) {
    union { unsigned short s; __bf16 b; } c; c.s = s; return c.b;
}
DEV __bf16 f2bf(float f) { return bfbits(f2bfbits(f)); }

// ---------------- cross-lane reductions (wave32) ----------------
DEV float sum16(float v) {
    #pragma unroll
    for (int m = 1; m <= 8; m <<= 1) v += __shfl_xor(v, m, 32);
    return v;
}
DEV float max16(float v) {
    #pragma unroll
    for (int m = 1; m <= 8; m <<= 1) v = fmaxf(v, __shfl_xor(v, m, 32));
    return v;
}
DEV float sum32(float v) {
    #pragma unroll
    for (int m = 1; m <= 16; m <<= 1) v += __shfl_xor(v, m, 32);
    return v;
}

// K-index interleave for 16-bit A/B fragments of v_wmma_*_16x16x32 (ISA 7.12.2)
DEV int kidx(int t, int h) { return ((t < 4) ? 0 : 16) + h * 8 + 2 * (t & 3); }

// =====================================================================
// 1) rank within label group (replaces stable argsort) + per-part counts
// =====================================================================
__global__ void rank_kernel(const int* __restrict__ lbl, const float* __restrict__ ru,
                            int* __restrict__ rank, int* __restrict__ counts)
{
    __shared__ int   sl[NIN];
    __shared__ float sr[NIN];
    __shared__ int   sc[PART];
    int b = blockIdx.x, tid = threadIdx.x;
    for (int i = tid; i < NIN; i += 256) { sl[i] = lbl[b * NIN + i]; sr[i] = ru[b * NIN + i]; }
    if (tid < PART) sc[tid] = 0;
    __syncthreads();
    for (int i = tid; i < NIN; i += 256) {
        int li = sl[i]; float ri = sr[i]; int r = 0;
        for (int j = 0; j < NIN; ++j) {
            if (sl[j] == li && (sr[j] < ri || (sr[j] == ri && j < i))) ++r;
        }
        rank[b * NIN + i] = r;
        atomicAdd(&sc[li], 1);
    }
    __syncthreads();
    if (tid < PART) counts[b * PART + tid] = sc[tid];
}

// =====================================================================
// 2) per-batch scalars: part_select, offsets, cnt, cdf, tot flag
// =====================================================================
__global__ void params_kernel(const float* __restrict__ pr, const float* __restrict__ ir,
                              const int* __restrict__ nv, const int* __restrict__ counts,
                              int* __restrict__ psel, int* __restrict__ offs,
                              int* __restrict__ cnt, float* __restrict__ cdf,
                              int* __restrict__ totok)
{
    int b = blockIdx.x;
    if (threadIdx.x != 0) return;
    float nf = (float)(nv[b] - (MAX_TOKEN - 2));
    float prob[PART]; float tot = 0.f; int running = 0;
    for (int j = 0; j < PART; ++j) {
        float pj  = pr[b * PART + j];
        float pp  = rintf(pj * (float)P);
        float pad = rintf(pj * (float)P - ir[b * PART + j] * nf);
        float padp = fmaxf(pad, 0.f);
        int ps = (int)fmaxf(pp - padp, 0.f);
        int cj = counts[b * PART + j];
        int tk = (ps < cj) ? ps : cj;
        psel[b * PART + j] = ps;
        offs[b * PART + j] = running;
        running += tk;
        prob[j] = padp; tot += padp;
    }
    cnt[b] = running;
    float acc = 0.f, dn = fmaxf(tot, 1e-9f);
    for (int j = 0; j < PART; ++j) { acc += prob[j]; cdf[b * PART + j] = acc / dn; }
    totok[b] = (tot >= 1.0f) ? 1 : 0;
}

// =====================================================================
// 3) init balls/labels/mask, scatter, multinomial fill
// =====================================================================
__global__ void init_kernel(const float* __restrict__ binit, float* __restrict__ balls,
                            int* __restrict__ labels, int* __restrict__ mask)
{
    int i = blockIdx.x * 256 + threadIdx.x;           // BS*P
    balls[(size_t)i * 3 + 0] = binit[(size_t)i * 3 + 0];
    balls[(size_t)i * 3 + 1] = binit[(size_t)i * 3 + 1];
    balls[(size_t)i * 3 + 2] = binit[(size_t)i * 3 + 2];
    labels[i] = 0; mask[i] = 0;
}

__global__ void scatter_kernel(const float* __restrict__ pt, const int* __restrict__ lbl,
                               const int* __restrict__ rank, const int* __restrict__ psel,
                               const int* __restrict__ offs, float* __restrict__ balls,
                               int* __restrict__ labels, int* __restrict__ mask)
{
    int g = blockIdx.x * 256 + threadIdx.x;           // BS*NIN
    int b = g / NIN;
    int li = lbl[g]; int r = rank[g];
    if (r >= psel[b * PART + li]) return;
    int slot = offs[b * PART + li] + r;
    if (slot >= P) return;
    size_t d = (size_t)(b * P + slot);
    balls[d * 3 + 0] = pt[(size_t)g * 3 + 0];
    balls[d * 3 + 1] = pt[(size_t)g * 3 + 1];
    balls[d * 3 + 2] = pt[(size_t)g * 3 + 2];
    labels[d] = li; mask[d] = 1;
}

__global__ void fill_kernel(const float* __restrict__ rf, const int* __restrict__ cnt,
                            const float* __restrict__ cdf, const int* __restrict__ totok,
                            int* __restrict__ labels)
{
    int g = blockIdx.x * 256 + threadIdx.x;           // BS*P
    int b = g >> 10, p = g & (P - 1);
    if (p < cnt[b] || !totok[b]) return;
    float rv = rf[g];
    int f = 0;
    #pragma unroll
    for (int j = 0; j < PART; ++j) f += (rv > cdf[b * PART + j]) ? 1 : 0;
    labels[g] = f;
}

// =====================================================================
// 4) embedding: proj[b,part,:] = enc[b,part,:] @ W_emb[3:259], then gather
// =====================================================================
__global__ void proj_kernel(const float* __restrict__ enc, const float* __restrict__ Wemb,
                            float* __restrict__ proj)
{
    int bp = blockIdx.x;                               // BS*PART
    int o = threadIdx.x;                               // 256
    const float* er = enc + (size_t)bp * HID;
    float s = 0.f;
    for (int c = 0; c < HID; ++c) s += er[c] * Wemb[(size_t)(3 + c) * HID + o];
    proj[(size_t)bp * HID + o] = s;
}

__global__ void embed_kernel(const float* __restrict__ balls, const int* __restrict__ labels,
                             const int* __restrict__ mask, const float* __restrict__ proj,
                             const float* __restrict__ Wemb, const float* __restrict__ bemb,
                             const float* __restrict__ lemb, float* __restrict__ x)
{
    int tok = blockIdx.x;                              // BS*P
    int o = threadIdx.x;                               // 256
    int b = tok >> 10;
    int lab = labels[tok]; int mk = mask[tok];
    const float* bl = balls + (size_t)tok * 3;
    float s = bl[0] * Wemb[o] + bl[1] * Wemb[HID + o] + bl[2] * Wemb[2 * HID + o];
    s += proj[(size_t)(b * PART + lab) * HID + o];
    s += Wemb[(size_t)(259 + mk) * HID + o];           // one_hot(mask,2) rows 259/260
    s += bemb[o] + lemb[(size_t)lab * HID + o];
    x[(size_t)tok * HID + o] = s;
}

// =====================================================================
// 5) LayerNorm (affine), f32 in -> bf16 out; one wave per token
// =====================================================================
__global__ void ln_kernel(const float* __restrict__ x, const float* __restrict__ g,
                          const float* __restrict__ bta, unsigned short* __restrict__ out)
{
    int wave = threadIdx.x >> 5, lane = threadIdx.x & 31;
    int tok = blockIdx.x * 8 + wave;
    const float* xr = x + (size_t)tok * HID;
    float vals[8], s = 0.f;
    #pragma unroll
    for (int t = 0; t < 8; ++t) { vals[t] = xr[lane + 32 * t]; s += vals[t]; }
    float mean = sum32(s) * (1.0f / HID);
    float vs = 0.f;
    #pragma unroll
    for (int t = 0; t < 8; ++t) { float d = vals[t] - mean; vs += d * d; }
    float inv = rsqrtf(sum32(vs) * (1.0f / HID) + 1e-5f);
    #pragma unroll
    for (int t = 0; t < 8; ++t) {
        int c = lane + 32 * t;
        out[(size_t)tok * HID + c] = f2bfbits((vals[t] - mean) * inv * g[c] + bta[c]);
    }
}

// =====================================================================
// 6) WMMA GEMM:  out = [relu]( A(bf16)[M,K] @ B(f32)[K,N] + bias [+ resid] )
//    Block = 256-row x 16-col strip: 8 waves x two 16x16 m-tiles.
//    B tile (32x16 per K-step) staged once through LDS (stored [n][k] so each
//    wave rebuilds its fragment with contiguous ds_load_b32), 16x global reuse.
// =====================================================================
template<bool OBF>
__global__ __launch_bounds__(256)
void gemm_ws(const unsigned short* __restrict__ A, const float* __restrict__ B,
             const float* __restrict__ bias, const float* __restrict__ resid,
             void* __restrict__ out, int M, int N, int K, int relu)
{
    __shared__ unsigned short bt[16 * 32];            // [n][k] transposed B tile
    int wave = threadIdx.x >> 5, lane = threadIdx.x & 31;
    int tiles_n = N >> 4;
    int bm = blockIdx.x / tiles_n;                    // 256-row strip
    int tn = blockIdx.x - bm * tiles_n;
    int m0 = bm * 256 + wave * 32;                    // two m-tiles: m0, m0+16
    int n0 = tn << 4;
    int m = lane & 15, hl = lane >> 4;

    // staging role: thread -> (col sn, k-pair sk2)
    int sn  = threadIdx.x & 15;
    int sk2 = threadIdx.x >> 4;                       // 0..15 -> k = 2*sk2, 2*sk2+1

    v8f c0, c1;
    #pragma unroll
    for (int i = 0; i < 8; ++i) { c0[i] = 0.f; c1[i] = 0.f; }

    for (int k0 = 0; k0 < K; k0 += 32) {
        // ---- stage B tile into LDS (f32 -> bf16, transposed) ----
        float f0 = B[(size_t)(k0 + 2 * sk2) * N + n0 + sn];
        float f1 = B[(size_t)(k0 + 2 * sk2 + 1) * N + n0 + sn];
        if (k0 + 32 < K)                               // prefetch next B k-rows
            __builtin_prefetch(&B[(size_t)(k0 + 32 + 2 * sk2) * N + n0 + sn], 0, 1);
        __syncthreads();                               // prior reads of bt done
        unsigned int packed = (unsigned int)f2bfbits(f0) | ((unsigned int)f2bfbits(f1) << 16);
        *(unsigned int*)&bt[sn * 32 + 2 * sk2] = packed;
        __syncthreads();

        // ---- B fragment from LDS: contiguous 32-bit reads ----
        v16bf bf;
        #pragma unroll
        for (int t = 0; t < 8; ++t) {
            int kk = kidx(t, hl);
            unsigned int w = *(const unsigned int*)&bt[(lane & 15) * 32 + kk];
            bf[2 * t]     = bfbits((unsigned short)(w & 0xffffu));
            bf[2 * t + 1] = bfbits((unsigned short)(w >> 16));
        }

        // ---- A fragments (two m-tiles) + WMMA ----
        const unsigned short* arow0 = A + (size_t)(m0 + m) * K + k0;
        const unsigned short* arow1 = arow0 + (size_t)16 * K;
        v16bf a0, a1;
        #pragma unroll
        for (int t = 0; t < 8; ++t) {
            int kk = kidx(t, hl);
            unsigned int w0 = *(const unsigned int*)(arow0 + kk);
            unsigned int w1 = *(const unsigned int*)(arow1 + kk);
            a0[2 * t]     = bfbits((unsigned short)(w0 & 0xffffu));
            a0[2 * t + 1] = bfbits((unsigned short)(w0 >> 16));
            a1[2 * t]     = bfbits((unsigned short)(w1 & 0xffffu));
            a1[2 * t + 1] = bfbits((unsigned short)(w1 >> 16));
        }
        c0 = __builtin_amdgcn_wmma_f32_16x16x32_bf16(false, a0, false, bf, (short)0, c0, false, false);
        c1 = __builtin_amdgcn_wmma_f32_16x16x32_bf16(false, a1, false, bf, (short)0, c1, false, false);
    }

    int col = n0 + (lane & 15);
    float bv = bias[col];
    #pragma unroll
    for (int r = 0; r < 8; ++r) {
        int row0 = m0 + r + 8 * hl;
        int row1 = row0 + 16;
        float v0 = c0[r] + bv;
        float v1 = c1[r] + bv;
        if (resid) { v0 += resid[(size_t)row0 * N + col]; v1 += resid[(size_t)row1 * N + col]; }
        if (relu)  { v0 = fmaxf(v0, 0.f); v1 = fmaxf(v1, 0.f); }
        if (OBF) {
            ((unsigned short*)out)[(size_t)row0 * N + col] = f2bfbits(v0);
            ((unsigned short*)out)[(size_t)row1 * N + col] = f2bfbits(v1);
        } else {
            ((float*)out)[(size_t)row0 * N + col] = v0;
            ((float*)out)[(size_t)row1 * N + col] = v1;
        }
    }
}

// =====================================================================
// 7) fused flash attention: one wave per (batch, head, 16-query tile)
//    q,k,v,o are bf16 [BS*SEQ, HID]; head h occupies cols h*HD..h*HD+31
// =====================================================================
__global__ __launch_bounds__(256)
void attn_kernel(const unsigned short* __restrict__ q, const unsigned short* __restrict__ k,
                 const unsigned short* __restrict__ v, unsigned short* __restrict__ o)
{
    __shared__ unsigned short plds[8][16 * 32];
    int wave = threadIdx.x >> 5, lane = threadIdx.x & 31;
    int task = blockIdx.x * 8 + wave;       // BS*NH*(SEQ/16) = 16384
    int mt = task & 63;
    int hd = (task >> 6) & 7;
    int b  = task >> 9;
    int m0 = mt << 4;
    int m = lane & 15, hl = lane >> 4;
    const float scale = 0.1767766952966369f;   // 1/sqrt(32)

    // Q A-fragment (K = HD = 32 exactly one step)
    v16bf qa;
    const unsigned short* qrow = q + (size_t)(b * SEQ + m0 + m) * HID + hd * HD;
    #pragma unroll
    for (int t = 0; t < 8; ++t) {
        int kk = kidx(t, hl);
        unsigned int w = *(const unsigned int*)(qrow + kk);
        qa[2 * t]     = bfbits((unsigned short)(w & 0xffffu));
        qa[2 * t + 1] = bfbits((unsigned short)(w >> 16));
    }

    v8f acc0, acc1;
    float rmax[8], rsum[8];
    #pragma unroll
    for (int r = 0; r < 8; ++r) { acc0[r] = 0.f; acc1[r] = 0.f; rmax[r] = -3.0e38f; rsum[r] = 0.f; }

    for (int j0 = 0; j0 < SEQ; j0 += 32) {
        // K^T B-fragments: column n = key, elements kk = head-dim (contiguous loads)
        v16bf kb0, kb1;
        {
            const unsigned short* kr0 = k + (size_t)(b * SEQ + j0 + m) * HID + hd * HD;
            const unsigned short* kr1 = k + (size_t)(b * SEQ + j0 + 16 + m) * HID + hd * HD;
            #pragma unroll
            for (int t = 0; t < 8; ++t) {
                int kk = kidx(t, hl);
                unsigned int w0 = *(const unsigned int*)(kr0 + kk);
                unsigned int w1 = *(const unsigned int*)(kr1 + kk);
                kb0[2 * t] = bfbits((unsigned short)(w0 & 0xffffu));
                kb0[2 * t + 1] = bfbits((unsigned short)(w0 >> 16));
                kb1[2 * t] = bfbits((unsigned short)(w1 & 0xffffu));
                kb1[2 * t + 1] = bfbits((unsigned short)(w1 >> 16));
            }
        }
        v8f s0, s1;
        #pragma unroll
        for (int i = 0; i < 8; ++i) { s0[i] = 0.f; s1[i] = 0.f; }
        s0 = __builtin_amdgcn_wmma_f32_16x16x32_bf16(false, qa, false, kb0, (short)0, s0, false, false);
        s1 = __builtin_amdgcn_wmma_f32_16x16x32_bf16(false, qa, false, kb1, (short)0, s1, false, false);

        // online softmax update per row (row r+8*hl lives across the 16-lane half)
        #pragma unroll
        for (int r = 0; r < 8; ++r) {
            float a0 = s0[r] * scale, a1 = s1[r] * scale;
            float cm = max16(fmaxf(a0, a1));
            float mn = fmaxf(rmax[r], cm);
            float p0 = __expf(a0 - mn), p1 = __expf(a1 - mn);
            float ps = sum16(p0 + p1);
            float es = __expf(rmax[r] - mn);
            rsum[r] = rsum[r] * es + ps;
            rmax[r] = mn;
            acc0[r] *= es; acc1[r] *= es;
            int row = r + 8 * hl, c0 = lane & 15;
            plds[wave][row * 32 + c0]      = f2bfbits(p0);
            plds[wave][row * 32 + 16 + c0] = f2bfbits(p1);
        }
        __syncthreads();

        // P A-fragment from LDS (C-layout -> A-layout transpose)
        v16bf pa;
        #pragma unroll
        for (int t = 0; t < 8; ++t) {
            int kk = kidx(t, hl);
            unsigned int w = *(const unsigned int*)&plds[wave][m * 32 + kk];
            pa[2 * t]     = bfbits((unsigned short)(w & 0xffffu));
            pa[2 * t + 1] = bfbits((unsigned short)(w >> 16));
        }

        // V B-fragments: column n = output dim, elements kk = key (stride HID)
        v16bf vb0, vb1;
        #pragma unroll
        for (int t = 0; t < 8; ++t) {
            int kk = kidx(t, hl);
            const unsigned short* vr0 = v + (size_t)(b * SEQ + j0 + kk) * HID + hd * HD;
            const unsigned short* vr1 = v + (size_t)(b * SEQ + j0 + kk + 1) * HID + hd * HD;
            int c0 = lane & 15;
            vb0[2 * t]     = bfbits(vr0[c0]);
            vb0[2 * t + 1] = bfbits(vr1[c0]);
            vb1[2 * t]     = bfbits(vr0[16 + c0]);
            vb1[2 * t + 1] = bfbits(vr1[16 + c0]);
        }
        acc0 = __builtin_amdgcn_wmma_f32_16x16x32_bf16(false, pa, false, vb0, (short)0, acc0, false, false);
        acc1 = __builtin_amdgcn_wmma_f32_16x16x32_bf16(false, pa, false, vb1, (short)0, acc1, false, false);
        __syncthreads();
    }

    #pragma unroll
    for (int r = 0; r < 8; ++r) {
        float inv = 1.0f / rsum[r];
        size_t row = (size_t)(b * SEQ + m0 + r + 8 * hl);
        int c0 = lane & 15;
        o[row * HID + hd * HD + c0]      = f2bfbits(acc0[r] * inv);
        o[row * HID + hd * HD + 16 + c0] = f2bfbits(acc1[r] * inv);
    }
}

// =====================================================================
// 8) MLP head: 256 ->(LN,relu) 128 ->(LN,relu) 32 -> 3, + balls
// =====================================================================
__global__ void head_kernel(const unsigned short* __restrict__ h,
                            const float* __restrict__ Wf1, const float* __restrict__ bf1v,
                            const float* __restrict__ Wf2, const float* __restrict__ bf2v,
                            const float* __restrict__ Wf3, const float* __restrict__ bf3v,
                            const float* __restrict__ balls, float* __restrict__ out)
{
    __shared__ float hs[8][HID];
    __shared__ float y1s[8][128];
    int wave = threadIdx.x >> 5, lane = threadIdx.x & 31;
    int tok = blockIdx.x * 8 + wave;
    const unsigned short* hr = h + (size_t)tok * HID;
    #pragma unroll
    for (int t = 0; t < 8; ++t) hs[wave][lane + 32 * t] = bf2f(hr[lane + 32 * t]);
    __syncthreads();

    float y1[4], s4 = 0.f;
    #pragma unroll
    for (int t = 0; t < 4; ++t) {
        int o = lane + 32 * t;
        float s = bf1v[o];
        for (int c = 0; c < HID; ++c) s += hs[wave][c] * Wf1[(size_t)c * 128 + o];
        y1[t] = s; s4 += s;
    }
    float mean = sum32(s4) * (1.0f / 128.0f);
    float v4 = 0.f;
    #pragma unroll
    for (int t = 0; t < 4; ++t) { float d = y1[t] - mean; v4 += d * d; }
    float inv = rsqrtf(sum32(v4) * (1.0f / 128.0f) + 1e-5f);
    #pragma unroll
    for (int t = 0; t < 4; ++t)
        y1s[wave][lane + 32 * t] = fmaxf((y1[t] - mean) * inv, 0.f);
    __syncthreads();

    float s2 = bf2v[lane];
    for (int c = 0; c < 128; ++c) s2 += y1s[wave][c] * Wf2[(size_t)c * 32 + lane];
    float m2 = sum32(s2) * (1.0f / 32.0f);
    float d2 = s2 - m2;
    float inv2 = rsqrtf(sum32(d2 * d2) * (1.0f / 32.0f) + 1e-5f);
    float y2 = fmaxf(d2 * inv2, 0.f);

    float o0 = sum32(y2 * Wf3[lane * 3 + 0]);
    float o1 = sum32(y2 * Wf3[lane * 3 + 1]);
    float o2 = sum32(y2 * Wf3[lane * 3 + 2]);
    if (lane == 0) {
        const float* bl = balls + (size_t)tok * 3;
        out[(size_t)tok * 3 + 0] = o0 + bf3v[0] + bl[0];
        out[(size_t)tok * 3 + 1] = o1 + bf3v[1] + bl[1];
        out[(size_t)tok * 3 + 2] = o2 + bf3v[2] + bl[2];
    }
}

__global__ void labels_out_kernel(const int* __restrict__ labels, float* __restrict__ out)
{
    int g = blockIdx.x * 256 + threadIdx.x;
    out[g] = (float)labels[g];
}

// =====================================================================
// host launcher
// =====================================================================
extern "C" void kernel_launch(void* const* d_in, const int* in_sizes, int n_in,
                              void* d_out, int out_size, void* d_ws, size_t ws_size,
                              hipStream_t stream)
{
    (void)in_sizes; (void)n_in; (void)out_size; (void)ws_size;
    // input order: setup_inputs dict, params flattened in insertion order
    const float* enc        = (const float*)d_in[0];
    const float* input_pt   = (const float*)d_in[1];
    const float* input_rat  = (const float*)d_in[2];
    const float* point_rat  = (const float*)d_in[3];
    const float* rand_u     = (const float*)d_in[4];
    const float* rand_fill  = (const float*)d_in[5];
    const float* balls_init = (const float*)d_in[6];
    const float* W_emb  = (const float*)d_in[7];
    const float* b_emb  = (const float*)d_in[8];
    const float* l_emb  = (const float*)d_in[9];
    const float* ln1_g  = (const float*)d_in[10];
    const float* ln1_b  = (const float*)d_in[11];
    const float* ln2_g  = (const float*)d_in[12];
    const float* ln2_b  = (const float*)d_in[13];
    const float* Wq = (const float*)d_in[14]; const float* bq = (const float*)d_in[15];
    const float* Wk = (const float*)d_in[16]; const float* bk = (const float*)d_in[17];
    const float* Wv = (const float*)d_in[18]; const float* bv = (const float*)d_in[19];
    const float* Wo = (const float*)d_in[20]; const float* bo = (const float*)d_in[21];
    const float* W1 = (const float*)d_in[22]; const float* b1 = (const float*)d_in[23];
    const float* W2 = (const float*)d_in[24]; const float* b2 = (const float*)d_in[25];
    const float* lnf_g = (const float*)d_in[26];
    const float* lnf_b = (const float*)d_in[27];
    const float* Wf1 = (const float*)d_in[28]; const float* bf1 = (const float*)d_in[29];
    const float* Wf2 = (const float*)d_in[30]; const float* bf2 = (const float*)d_in[31];
    const float* Wf3 = (const float*)d_in[32]; const float* bf3 = (const float*)d_in[33];
    const int* input_label = (const int*)d_in[34];
    const int* nvals       = (const int*)d_in[35];

    // workspace carve-up
    char* ws = (char*)d_ws;
    size_t off = 0;
    auto take = [&](size_t bytes) -> void* {
        void* p = ws + off; off = (off + bytes + 255) & ~(size_t)255; return p;
    };
    int*   counts = (int*)  take(BS * PART * 4);
    int*   psel   = (int*)  take(BS * PART * 4);
    int*   offs   = (int*)  take(BS * PART * 4);
    int*   cntb   = (int*)  take(BS * 4);
    float* cdf    = (float*)take(BS * PART * 4);
    int*   totok  = (int*)  take(BS * 4);
    int*   rank   = (int*)  take((size_t)BS * NIN * 4);
    int*   labels = (int*)  take((size_t)BS * P * 4);
    int*   mask   = (int*)  take((size_t)BS * P * 4);
    float* balls  = (float*)take((size_t)BS * P * 3 * 4);
    float* proj   = (float*)take((size_t)BS * PART * HID * 4);
    float* x      = (float*)take((size_t)BS * P * HID * 4);            // 32 MB
    unsigned short* hbuf = (unsigned short*)take((size_t)BS * P * HID * 2);   // 16 MB
    unsigned short* qkvo = (unsigned short*)take((size_t)BS * P * HID * 2 * 4); // 64 MB
    unsigned short* qb = qkvo;
    unsigned short* kb = qkvo + (size_t)BS * P * HID;
    unsigned short* vb = qkvo + (size_t)2 * BS * P * HID;
    unsigned short* ob = qkvo + (size_t)3 * BS * P * HID;
    unsigned short* fb = qkvo;                                         // FFN hidden aliases q/k/v/o

    const int M = BS * P;                 // 32768 tokens
    const int TOKB = M / 8;               // 4096 blocks (one wave per token)

    // -------- preamble --------
    rank_kernel   <<<BS, 256, 0, stream>>>(input_label, rand_u, rank, counts);
    params_kernel <<<BS, 32, 0, stream>>>(point_rat, input_rat, nvals, counts,
                                          psel, offs, cntb, cdf, totok);
    init_kernel   <<<M / 256, 256, 0, stream>>>(balls_init, balls, labels, mask);
    scatter_kernel<<<BS * NIN / 256, 256, 0, stream>>>(input_pt, input_label, rank,
                                                       psel, offs, balls, labels, mask);
    fill_kernel   <<<M / 256, 256, 0, stream>>>(rand_fill, cntb, cdf, totok, labels);

    // -------- embedding --------
    proj_kernel  <<<BS * PART, 256, 0, stream>>>(enc, W_emb, proj);
    embed_kernel <<<M, 256, 0, stream>>>(balls, labels, mask, proj, W_emb, b_emb, l_emb, x);

    // -------- transformer layers --------
    const int G256  = (M / 256) * (HID / 16);      // 2048 blocks (N=256 GEMM)
    const int G1024 = (M / 256) * (FFD / 16);      // 8192 blocks (N=1024 GEMM)
    const int GATT  = BS * NH * (SEQ / 16) / 8;    // 2048 blocks

    for (int l = 0; l < NL; ++l) {
        ln_kernel<<<TOKB, 256, 0, stream>>>(x, ln1_g + l * HID, ln1_b + l * HID, hbuf);
        gemm_ws<true> <<<G256, 256, 0, stream>>>(hbuf, Wq + (size_t)l * HID * HID, bq + l * HID,
                                                 nullptr, qb, M, HID, HID, 0);
        gemm_ws<true> <<<G256, 256, 0, stream>>>(hbuf, Wk + (size_t)l * HID * HID, bk + l * HID,
                                                 nullptr, kb, M, HID, HID, 0);
        gemm_ws<true> <<<G256, 256, 0, stream>>>(hbuf, Wv + (size_t)l * HID * HID, bv + l * HID,
                                                 nullptr, vb, M, HID, HID, 0);
        attn_kernel   <<<GATT, 256, 0, stream>>>(qb, kb, vb, ob);
        gemm_ws<false><<<G256, 256, 0, stream>>>(ob, Wo + (size_t)l * HID * HID, bo + l * HID,
                                                 x, x, M, HID, HID, 0);
        ln_kernel<<<TOKB, 256, 0, stream>>>(x, ln2_g + l * HID, ln2_b + l * HID, hbuf);
        gemm_ws<true> <<<G1024, 256, 0, stream>>>(hbuf, W1 + (size_t)l * HID * FFD, b1 + l * FFD,
                                                  nullptr, fb, M, FFD, HID, 1);
        gemm_ws<false><<<G256, 256, 0, stream>>>(fb, W2 + (size_t)l * FFD * HID, b2 + l * HID,
                                                 x, x, M, HID, FFD, 0);
    }

    // -------- final LN + head --------
    ln_kernel  <<<TOKB, 256, 0, stream>>>(x, lnf_g, lnf_b, hbuf);
    head_kernel<<<TOKB, 256, 0, stream>>>(hbuf, Wf1, bf1, Wf2, bf2, Wf3, bf3,
                                          balls, (float*)d_out);
    labels_out_kernel<<<M / 256, 256, 0, stream>>>(labels, (float*)d_out + (size_t)M * 3);
}
